// AttnFree_Block_82291573391767
// MI455X (gfx1250) — matrix operand order repaired and checked by reading, not compiled, not run on hardware
//
#include <hip/hip_runtime.h>
#include <hip/hip_bf16.h>
#include <math.h>

typedef __attribute__((ext_vector_type(16))) _Float16 v16h;
typedef __attribute__((ext_vector_type(8)))  _Float16 v8h;
typedef __attribute__((ext_vector_type(8)))  float    v8f;

#define NB   2
#define NT   512
#define ND   128
#define NH   128
#define NHID 64
#define NFF  512

// ---------------------------------------------------------------------------
// WMMA helpers (CDNA5 wave32, 16x16x32 f16 -> f32)
// ---------------------------------------------------------------------------
__device__ __forceinline__ v8f wmma16(v16h a, v16h b, v8f c) {
  // (neg_a, A, neg_b, B, c_mod, C, reuse_a, reuse_b)
  return __builtin_amdgcn_wmma_f32_16x16x32_f16(false, a, false, b, (short)0, c,
                                                false, false);
}

// A-fragment for a 16x32 f16 tile, row-major LDS row of >=32 halves.
// Layout: lanes 0-15 hold K[0..15] contiguous;
//         lanes 16-31 hold K[16..31] contiguous.
__device__ __forceinline__ v16h ld_afrag(const _Float16* row, int hsel) {
  v8h lo = *(const v8h*)(row + hsel * 16);
  v8h hi = *(const v8h*)(row + hsel * 16 + 8);
  v16h r;
#pragma unroll
  for (int i = 0; i < 8; ++i) { r[i] = lo[i]; r[8 + i] = hi[i]; }
  return r;
}

// B-fragment for a 32x16 f16 tile from an [N][K]-transposed LDS array:
// lane L(0-15): col N=L, K=0..15 contiguous; lanes 16-31: same col, K=16..31.
__device__ __forceinline__ v16h ld_bfrag(const _Float16* colK) {
  return *(const v16h*)colK;
}

__device__ __forceinline__ v8f zero8() {
  v8f a;
#pragma unroll
  for (int r = 0; r < 8; ++r) a[r] = 0.f;
  return a;
}

// Fast reciprocal -> single v_rcp_f32 (avoids IEEE div_scale/fma expansion).
__device__ __forceinline__ float frcp(float v) { return __builtin_amdgcn_rcpf(v); }
__device__ __forceinline__ float siluf(float v) { return v * frcp(1.f + __expf(-v)); }
__device__ __forceinline__ float sigmf(float v) { return frcp(1.f + __expf(-v)); }

// ---------------------------------------------------------------------------
// Stage 1: fused pairwise adaptive-bias network.
// Block = 128 threads = 4 waves; each wave handles 16 (i,j) pairs.
// All GEMMs (9->64, 64->64 x2 channels, 128->64 gate) are f16 WMMA.
// ---------------------------------------------------------------------------
__global__ __launch_bounds__(128)
void stage1_bias_kernel(const float* __restrict__ coords,
                        const float* __restrict__ cost_mat,
                        const float* __restrict__ log_scale,
                        const float* __restrict__ mlp_w1, const float* __restrict__ mlp_b1,
                        const float* __restrict__ mlp_w2, const float* __restrict__ mlp_b2,
                        const float* __restrict__ film_g, const float* __restrict__ film_b,
                        const float* __restrict__ gate_w1, const float* __restrict__ gate_b1,
                        const float* __restrict__ gate_w2, const float* __restrict__ gate_b2,
                        const float* __restrict__ gate_temp,
                        const float* __restrict__ ln_g, const float* __restrict__ ln_b,
                        const float* __restrict__ proj_w, const float* __restrict__ proj_b,
                        float* __restrict__ bias_out)
{
  __shared__ __align__(16) _Float16 w1t[64][32];    // mlp_w1^T, K padded 9->32
  __shared__ __align__(16) _Float16 w2t[64][64];    // mlp_w2^T
  __shared__ __align__(16) _Float16 gw1t[64][128];  // gate_w1^T
  __shared__ __align__(16) _Float16 h1f[4][16][64];     // per-wave hidden
  __shared__ __align__(16) _Float16 gatein[4][16][128]; // per-wave [e0|e1]
  __shared__ float sb1[64], sb2[64], sgb1[64];
  __shared__ float sfg[2][64], sfb[2][64], sgw2[2][64];
  __shared__ float slng[64], slnb[64], sproj[64];

  const int tid  = threadIdx.x;
  const int w    = tid >> 5;
  const int lane = tid & 31;
  const int rsel = lane & 15;   // row within 16-pair tile / col within n-tile
  const int hsel = lane >> 4;   // lane-group

  // --- stage shared weights (transposed to [N][K]) ---
  for (int idx = tid; idx < 64 * 32; idx += 128) {
    int n = idx >> 5, k = idx & 31;
    w1t[n][k] = (k < 9) ? (_Float16)mlp_w1[k * 64 + n] : (_Float16)0.f;
  }
  for (int idx = tid; idx < 64 * 64; idx += 128) {
    int n = idx >> 6, k = idx & 63;
    w2t[n][k] = (_Float16)mlp_w2[k * 64 + n];
  }
  for (int idx = tid; idx < 64 * 128; idx += 128) {
    int n = idx >> 7, k = idx & 127;
    gw1t[n][k] = (_Float16)gate_w1[k * 64 + n];
  }
  if (tid < 64) {
    sb1[tid] = mlp_b1[tid]; sb2[tid] = mlp_b2[tid]; sgb1[tid] = gate_b1[tid];
    slng[tid] = ln_g[tid];  slnb[tid] = ln_b[tid];  sproj[tid] = proj_w[tid];
    sfg[0][tid] = film_g[tid];      sfg[1][tid] = film_g[64 + tid];
    sfb[0][tid] = film_b[tid];      sfb[1][tid] = film_b[64 + tid];
    sgw2[0][tid] = gate_w2[tid * 2 + 0];
    sgw2[1][tid] = gate_w2[tid * 2 + 1];
  }
  __syncthreads();

  // --- per-pair Fourier features -> register A fragments (K padded to 32) ---
  const int p    = blockIdx.x * 128 + w * 16 + rsel;   // flat (b,i,j)
  const int bidx = p >> 18;                            // / (T*T)
  const int rem  = p & (NT * NT - 1);
  const int i    = rem >> 9;
  const int j    = rem & (NT - 1);

  const float s0   = __expf(log_scale[0]);
  const float s1   = __expf(log_scale[1]);
  const float cost = cost_mat[p];
  const float cix = coords[(bidx * NT + i) * 2 + 0];
  const float ciy = coords[(bidx * NT + i) * 2 + 1];
  const float cjx = coords[(bidx * NT + j) * 2 + 0];
  const float cjy = coords[(bidx * NT + j) * 2 + 1];
  const float ang = atan2f(ciy - cjy, cix - cjx);

  v16h afeat[2];
  {
    float fe[2][9];
    fe[0][0] = cost; fe[1][0] = ang;
    {
      float x = cost;
      fe[0][1] = __sinf(x);       fe[0][2] = __cosf(x);
      fe[0][3] = __sinf(2.f * x); fe[0][4] = __cosf(2.f * x);
      fe[0][5] = __sinf(4.f * x); fe[0][6] = __cosf(4.f * x);
      fe[0][7] = __sinf(8.f * x); fe[0][8] = __cosf(8.f * x);
    }
    {
      float x = ang;
      fe[1][1] = __sinf(x);       fe[1][2] = __cosf(x);
      fe[1][3] = __sinf(2.f * x); fe[1][4] = __cosf(2.f * x);
      fe[1][5] = __sinf(4.f * x); fe[1][6] = __cosf(4.f * x);
      fe[1][7] = __sinf(8.f * x); fe[1][8] = __cosf(8.f * x);
    }
    const float sc[2] = { s0, s1 };
#pragma unroll
    for (int c = 0; c < 2; ++c) {
#pragma unroll
      for (int t = 0; t < 16; ++t) afeat[c][t] = (_Float16)0.f;
      if (hsel == 0) {
#pragma unroll
        for (int t = 0; t < 9; ++t) afeat[c][t] = (_Float16)(fe[c][t] * sc[c]);
      }
    }
  }

  // --- two encoder channels (fully unrolled: afeat[c] stays in fixed VGPRs) ---
#pragma unroll
  for (int c = 0; c < 2; ++c) {
    // layer1: [16x32(9)] @ [32x64] -> SiLU -> h1f
#pragma unroll
    for (int nt = 0; nt < 4; ++nt) {
      v16h bf = ld_bfrag(&w1t[nt * 16 + rsel][hsel * 16]);
      v8f acc = zero8();
      acc = wmma16(afeat[c], bf, acc);
      int n = nt * 16 + rsel;
#pragma unroll
      for (int r = 0; r < 8; ++r) {
        float v = acc[r] + sb1[n];
        h1f[w][r + hsel * 8][n] = (_Float16)siluf(v);
      }
    }
    __syncthreads();

    // layer2: [16x64] @ [64x64] -> FiLM -> gatein[:, c*64 + n]
    v16h a2[2];
    a2[0] = ld_afrag(&h1f[w][rsel][0],  hsel);
    a2[1] = ld_afrag(&h1f[w][rsel][32], hsel);
#pragma unroll
    for (int nt = 0; nt < 4; ++nt) {
      v8f acc = zero8();
#pragma unroll
      for (int kc = 0; kc < 2; ++kc) {
        v16h bf = ld_bfrag(&w2t[nt * 16 + rsel][kc * 32 + hsel * 16]);
        acc = wmma16(a2[kc], bf, acc);
      }
      int n = nt * 16 + rsel;
#pragma unroll
      for (int r = 0; r < 8; ++r) {
        float v = acc[r] + sb2[n];
        v = v * sfg[c][n] + sfb[c][n];
        gatein[w][r + hsel * 8][c * 64 + n] = (_Float16)v;
      }
    }
    __syncthreads();
  }

  // --- gate layer1: [16x128] @ [128x64] -> SiLU -> h1f (reused) ---
  v16h ag[4];
#pragma unroll
  for (int kc = 0; kc < 4; ++kc) ag[kc] = ld_afrag(&gatein[w][rsel][kc * 32], hsel);
#pragma unroll
  for (int nt = 0; nt < 4; ++nt) {
    v8f acc = zero8();
#pragma unroll
    for (int kc = 0; kc < 4; ++kc) {
      v16h bf = ld_bfrag(&gw1t[nt * 16 + rsel][kc * 32 + hsel * 16]);
      acc = wmma16(ag[kc], bf, acc);
    }
    int n = nt * 16 + rsel;
#pragma unroll
    for (int r = 0; r < 8; ++r) {
      float v = acc[r] + sgb1[n];
      h1f[w][r + hsel * 8][n] = (_Float16)siluf(v);
    }
  }
  __syncthreads();

  // --- per-pair epilogue: gate head, softmax gate, LN, projection ---
  if (lane < 16) {
    float l0 = gate_b2[0], l1 = gate_b2[1];
    for (int k = 0; k < 64; ++k) {
      float g = (float)h1f[w][rsel][k];
      l0 += g * sgw2[0][k];
      l1 += g * sgw2[1][k];
    }
    const float it = __expf(-gate_temp[0]);
    l0 *= it; l1 *= it;
    float mx = fmaxf(l0, l1);
    float e0 = __expf(l0 - mx), e1 = __expf(l1 - mx);
    float g0 = e0 * frcp(e0 + e1), g1 = 1.f - g0;

    float sum = 0.f, ss = 0.f;
    for (int k = 0; k < 64; ++k) {
      float f = g0 * (float)gatein[w][rsel][k] + g1 * (float)gatein[w][rsel][64 + k];
      sum += f; ss += f * f;
    }
    float mu   = sum * (1.f / 64.f);
    float var  = ss * (1.f / 64.f) - mu * mu;
    float rstd = rsqrtf(var + 1e-5f);

    float adapt = proj_b[0];
    for (int k = 0; k < 64; ++k) {
      float f = g0 * (float)gatein[w][rsel][k] + g1 * (float)gatein[w][rsel][64 + k];
      adapt += ((f - mu) * rstd * slng[k] + slnb[k]) * sproj[k];
    }
    bias_out[p] = adapt;
  }
}

// ---------------------------------------------------------------------------
// Generic batched WMMA GEMM: C = act(A[MxK] @ B[KxN] + bias), f32 in/out,
// f16 compute, f32 accumulate. 64x64 tile per 128-thread block (4 waves).
// act: 0=none, 1=relu, 2=sigmoid. K must be a multiple of 32.
// ---------------------------------------------------------------------------
__global__ __launch_bounds__(128)
void gemm_kernel(const float* __restrict__ A, const float* __restrict__ B,
                 const float* __restrict__ bias, float* __restrict__ C,
                 int M, int N, int K, int act,
                 long long sA, long long sB, long long sC)
{
  __shared__ __align__(16) _Float16 Ah[64][32];  // row-major [m][k]
  __shared__ __align__(16) _Float16 Bt[64][32];  // transposed [n][k]

  const int tid  = threadIdx.x;
  const int w    = tid >> 5;
  const int lane = tid & 31;
  const int rsel = lane & 15;
  const int hsel = lane >> 4;
  const int m0   = blockIdx.y * 64;
  const int n0   = blockIdx.x * 64;

  A += (long long)blockIdx.z * sA;
  B += (long long)blockIdx.z * sB;
  C += (long long)blockIdx.z * sC;

  v8f acc[4];
#pragma unroll
  for (int t = 0; t < 4; ++t) acc[t] = zero8();

  for (int k0 = 0; k0 < K; k0 += 32) {
#pragma unroll
    for (int it = 0; it < 16; ++it) {           // stage A tile 64x32
      int idx = it * 128 + tid;
      int r = idx >> 5, c = idx & 31;
      int gm = m0 + r;
      Ah[r][c] = (gm < M) ? (_Float16)A[(long long)gm * K + (k0 + c)] : (_Float16)0.f;
    }
#pragma unroll
    for (int it = 0; it < 16; ++it) {           // stage B tile 32x64 transposed
      int idx = it * 128 + tid;
      int kk = idx >> 6, n = idx & 63;
      int gn = n0 + n;
      Bt[n][kk] = (gn < N) ? (_Float16)B[(long long)(k0 + kk) * N + gn] : (_Float16)0.f;
    }
    __syncthreads();

    v16h af = ld_afrag(&Ah[w * 16 + rsel][0], hsel);
#pragma unroll
    for (int nt = 0; nt < 4; ++nt) {
      v16h bf = ld_bfrag(&Bt[nt * 16 + rsel][hsel * 16]);
      acc[nt] = wmma16(af, bf, acc[nt]);
    }
    __syncthreads();
  }

#pragma unroll
  for (int nt = 0; nt < 4; ++nt) {
    int n = n0 + nt * 16 + rsel;
#pragma unroll
    for (int r = 0; r < 8; ++r) {
      int m = m0 + w * 16 + r + hsel * 8;
      if (m < M && n < N) {
        float v = acc[nt][r] + (bias ? bias[n] : 0.f);
        if (act == 1)      v = fmaxf(v, 0.f);
        else if (act == 2) v = sigmf(v);
        C[(long long)m * N + n] = v;
      }
    }
  }
}

// ---------------------------------------------------------------------------
// K token-softmax (axis=1) + eK, write concat [eK*V | eK] into ekcat (B,T,256)
// One block per (b,h).
// ---------------------------------------------------------------------------
__global__ __launch_bounds__(128)
void ksoftmax_kernel(const float* __restrict__ kraw, const float* __restrict__ vraw,
                     float* __restrict__ ekcat)
{
  __shared__ float red[128];
  const int bh = blockIdx.x;
  const int b = bh >> 7, h = bh & 127;
  const int tid = threadIdx.x;

  float vmax = -1e30f;
  for (int t = tid; t < NT; t += 128)
    vmax = fmaxf(vmax, kraw[(b * NT + t) * NH + h]);
  red[tid] = vmax; __syncthreads();
  for (int s = 64; s > 0; s >>= 1) { if (tid < s) red[tid] = fmaxf(red[tid], red[tid + s]); __syncthreads(); }
  vmax = red[0]; __syncthreads();

  float sum = 0.f;
  for (int t = tid; t < NT; t += 128)
    sum += __expf(kraw[(b * NT + t) * NH + h] - vmax);
  red[tid] = sum; __syncthreads();
  for (int s = 64; s > 0; s >>= 1) { if (tid < s) red[tid] += red[tid + s]; __syncthreads(); }
  const float inv = frcp(red[0]);

  for (int t = tid; t < NT; t += 128) {
    float ks = __expf(kraw[(b * NT + t) * NH + h] - vmax) * inv;   // softmax(K)
    float ek = __expf(ks);                                         // exp(softmax)
    ekcat[(b * NT + t) * (2 * NH) + h]      = ek * vraw[(b * NT + t) * NH + h];
    ekcat[(b * NT + t) * (2 * NH) + NH + h] = ek;
  }
}

// Row-wise A = exp(softmax(adapt_bias, axis=-1)) in place; one block per row.
__global__ __launch_bounds__(128)
void rowsoftmaxexp_kernel(float* __restrict__ a)
{
  __shared__ float red[128];
  const int row = blockIdx.x, tid = threadIdx.x;
  float v[4];
#pragma unroll
  for (int q = 0; q < 4; ++q) v[q] = a[(long long)row * NT + q * 128 + tid];

  float mx = fmaxf(fmaxf(v[0], v[1]), fmaxf(v[2], v[3]));
  red[tid] = mx; __syncthreads();
  for (int s = 64; s > 0; s >>= 1) { if (tid < s) red[tid] = fmaxf(red[tid], red[tid + s]); __syncthreads(); }
  mx = red[0]; __syncthreads();

  float sum = 0.f;
#pragma unroll
  for (int q = 0; q < 4; ++q) sum += __expf(v[q] - mx);
  red[tid] = sum; __syncthreads();
  for (int s = 64; s > 0; s >>= 1) { if (tid < s) red[tid] += red[tid + s]; __syncthreads(); }
  const float inv = frcp(red[0]);

#pragma unroll
  for (int q = 0; q < 4; ++q)
    a[(long long)row * NT + q * 128 + tid] = __expf(__expf(v[q] - mx) * inv);
}

// attin = Q_sig * num / den
__global__ __launch_bounds__(256)
void attcombine_kernel(const float* __restrict__ qsig, const float* __restrict__ numden,
                       float* __restrict__ attin, int n)
{
  int idx = blockIdx.x * 256 + threadIdx.x;
  if (idx >= n) return;
  int row = idx >> 7, h = idx & 127;
  float num = numden[row * (2 * NH) + h];
  float den = numden[row * (2 * NH) + NH + h];
  attin[idx] = qsig[idx] * num * frcp(den);
}

// out = rmsnorm(x + y, w), D = 128 = blockDim
__global__ __launch_bounds__(128)
void resrms_kernel(const float* __restrict__ x, const float* __restrict__ y,
                   const float* __restrict__ w, float* __restrict__ out)
{
  __shared__ float red[128];
  const int row = blockIdx.x, tid = threadIdx.x;
  float v = x[row * ND + tid] + y[row * ND + tid];
  red[tid] = v * v; __syncthreads();
  for (int s = 64; s > 0; s >>= 1) { if (tid < s) red[tid] += red[tid + s]; __syncthreads(); }
  float ms = red[0] * (1.f / (float)ND);
  out[row * ND + tid] = v * rsqrtf(ms + 1e-5f) * w[tid];
}

// ---------------------------------------------------------------------------
extern "C" void kernel_launch(void* const* d_in, const int* in_sizes, int n_in,
                              void* d_out, int out_size, void* d_ws, size_t ws_size,
                              hipStream_t stream) {
  const float* x         = (const float*)d_in[0];
  const float* coords    = (const float*)d_in[1];
  const float* cost_mat  = (const float*)d_in[2];
  const float* log_scale = (const float*)d_in[3];
  const float* mlp_w1    = (const float*)d_in[4];
  const float* mlp_b1    = (const float*)d_in[5];
  const float* mlp_w2    = (const float*)d_in[6];
  const float* mlp_b2    = (const float*)d_in[7];
  const float* film_g    = (const float*)d_in[8];
  const float* film_b    = (const float*)d_in[9];
  const float* gate_w1   = (const float*)d_in[10];
  const float* gate_b1   = (const float*)d_in[11];
  const float* gate_w2   = (const float*)d_in[12];
  const float* gate_b2   = (const float*)d_in[13];
  const float* gate_temp = (const float*)d_in[14];
  const float* ln_g      = (const float*)d_in[15];
  const float* ln_b      = (const float*)d_in[16];
  const float* proj_w    = (const float*)d_in[17];
  const float* proj_b    = (const float*)d_in[18];
  const float* wq = (const float*)d_in[19]; const float* bq = (const float*)d_in[20];
  const float* wk = (const float*)d_in[21]; const float* bk = (const float*)d_in[22];
  const float* wv = (const float*)d_in[23]; const float* bv = (const float*)d_in[24];
  const float* wo = (const float*)d_in[25]; const float* bo = (const float*)d_in[26];
  const float* rms1 = (const float*)d_in[27]; const float* rms2 = (const float*)d_in[28];
  const float* ff_w1 = (const float*)d_in[29]; const float* ff_b1 = (const float*)d_in[30];
  const float* ff_w2 = (const float*)d_in[31]; const float* ff_b2 = (const float*)d_in[32];
  float* out = (float*)d_out;

  float* ws = (float*)d_ws;
  float* biasA  = ws; ws += NB * NT * NT;       // adaptive bias, then A (in-place)
  float* qsig   = ws; ws += NB * NT * NH;
  float* kraw   = ws; ws += NB * NT * NH;
  float* vraw   = ws; ws += NB * NT * NH;
  float* ekcat  = ws; ws += NB * NT * 2 * NH;   // [eK*V | eK]
  float* numden = ws; ws += NB * NT * 2 * NH;   // [num | den]
  float* attin  = ws; ws += NB * NT * NH;
  float* atto   = ws; ws += NB * NT * ND;
  float* hbuf   = ws; ws += NB * NT * ND;
  float* ffn1   = ws; ws += NB * NT * NFF;
  float* ffn2   = ws; ws += NB * NT * ND;

  // ---- Stage 1: pairwise adaptive bias (WMMA-fused) ----
  stage1_bias_kernel<<<dim3(NB * NT * NT / 128), 128, 0, stream>>>(
      coords, cost_mat, log_scale, mlp_w1, mlp_b1, mlp_w2, mlp_b2,
      film_g, film_b, gate_w1, gate_b1, gate_w2, gate_b2, gate_temp,
      ln_g, ln_b, proj_w, proj_b, biasA);

  // ---- Stage 2: AFT ----
  gemm_kernel<<<dim3(NH / 64, (NB * NT) / 64, 1), 128, 0, stream>>>(
      x, wq, bq, qsig, NB * NT, NH, ND, 2, 0, 0, 0);
  gemm_kernel<<<dim3(NH / 64, (NB * NT) / 64, 1), 128, 0, stream>>>(
      x, wk, bk, kraw, NB * NT, NH, ND, 0, 0, 0, 0);
  gemm_kernel<<<dim3(NH / 64, (NB * NT) / 64, 1), 128, 0, stream>>>(
      x, wv, bv, vraw, NB * NT, NH, ND, 0, 0, 0, 0);

  ksoftmax_kernel<<<dim3(NB * NH), 128, 0, stream>>>(kraw, vraw, ekcat);
  rowsoftmaxexp_kernel<<<dim3(NB * NT), 128, 0, stream>>>(biasA);

  // [num|den] = A(B,T,T) @ ekcat(B,T,256), batched over B
  gemm_kernel<<<dim3((2 * NH) / 64, NT / 64, NB), 128, 0, stream>>>(
      biasA, ekcat, nullptr, numden, NT, 2 * NH, NT, 0,
      (long long)NT * NT, (long long)NT * 2 * NH, (long long)NT * 2 * NH);

  attcombine_kernel<<<dim3(NB * NT * NH / 256), 256, 0, stream>>>(
      qsig, numden, attin, NB * NT * NH);

  gemm_kernel<<<dim3(ND / 64, (NB * NT) / 64, 1), 128, 0, stream>>>(
      attin, wo, bo, atto, NB * NT, ND, NH, 0, 0, 0, 0);

  // ---- Stage 3: residual + RMSNorm + FFN ----
  resrms_kernel<<<dim3(NB * NT), 128, 0, stream>>>(x, atto, rms1, hbuf);
  gemm_kernel<<<dim3(NFF / 64, (NB * NT) / 64, 1), 128, 0, stream>>>(
      hbuf, ff_w1, ff_b1, ffn1, NB * NT, NFF, ND, 1, 0, 0, 0);
  gemm_kernel<<<dim3(ND / 64, (NB * NT) / 64, 1), 128, 0, stream>>>(
      ffn1, ff_w2, ff_b2, ffn2, NB * NT, ND, NFF, 0, 0, 0, 0);
  resrms_kernel<<<dim3(NB * NT), 128, 0, stream>>>(hbuf, ffn2, rms2, out);

  (void)in_sizes; (void)n_in; (void)out_size; (void)ws_size;
}
